// APP_74741020885554
// MI455X (gfx1250) — matrix operand order repaired and checked by reading, not compiled, not run on hardware
//
#include <hip/hip_runtime.h>
#include <cstdint>
#include <cstddef>

// ---------------------------------------------------------------------------
// Constants for this problem size
// ---------------------------------------------------------------------------
constexpr int NB    = 2;
constexpr int NCH   = 64;
constexpr int IMG_H = 256;
constexpr int IMG_W = 256;
constexpr int HW    = IMG_H * IMG_W;          // 65536
constexpr int WTILE = 128;                    // pixels per block (x)

typedef __attribute__((ext_vector_type(16))) __bf16 v16bf;
typedef __attribute__((ext_vector_type(8)))  float  v8f;

union FragBF {
    v16bf v;
    uint4 u[2];
    unsigned long long q[4];
};

__device__ __forceinline__ unsigned short f2bf(float f) {
    unsigned u = __float_as_uint(f);
    u += 0x7FFFu + ((u >> 16) & 1u);          // round-to-nearest-even
    return (unsigned short)(u >> 16);
}

// CDNA5 async global->LDS copy (ASYNCcnt-tracked). VDST = LDS byte address
// (low 32 bits of the flat shared pointer, per the LDS aperture rule),
// VADDR = 64-bit global address, saddr disabled.
__device__ __forceinline__ void async_g2l_b128(void* lds_dst, const void* gsrc) {
    unsigned lds32 = (unsigned)(uintptr_t)lds_dst;
    asm volatile("global_load_async_to_lds_b128 %0, %1, off"
                 :: "v"(lds32), "v"(gsrc) : "memory");
}
__device__ __forceinline__ void wait_asynccnt0() {
    asm volatile("s_wait_asynccnt 0" ::: "memory");
}

// ---------------------------------------------------------------------------
// Weight pre-pack: OIHW f32  ->  WMMA A-fragment bf16 layout
// frag[kt][mt(4)][lane(32)][e(16)]
// A 16x32 bf16 layout: lane l and l+16 hold row M=l.
//   lanes <16 : e=0..7 -> K=0..7,  e=8..15 -> K=16..23
//   lanes>=16 : e=0..7 -> K=8..15, e=8..15 -> K=24..31
// K-chunk order: kt = tap * nCc + cchunk  (tap-major, 32-channel chunks)
// ---------------------------------------------------------------------------
__global__ void pack_kernel(const float* __restrict__ w, unsigned short* __restrict__ frag,
                            int Cin, int Cout, int taps) {
    int nCc = (Cin + 31) / 32;
    long total = (long)taps * nCc * 4 * 32 * 16;
    long idx = (long)blockIdx.x * 256 + threadIdx.x;
    if (idx >= total) return;
    int e    = (int)(idx & 15);
    int lane = (int)((idx >> 4) & 31);
    int mt   = (int)((idx >> 9) & 3);
    int kt   = (int)(idx >> 11);
    int t  = kt / nCc;
    int cc = kt % nCc;
    int K  = (e < 8 ? e : e + 8) + ((lane >= 16) ? 8 : 0);   // 0..31
    int ci = cc * 32 + K;
    int co = mt * 16 + (lane & 15);
    float v = 0.f;
    if (co < Cout && ci < Cin) v = w[((long)co * Cin + ci) * taps + t];
    frag[idx] = f2bf(v);
}

// ---------------------------------------------------------------------------
// Implicit-GEMM convolution (templated on shape -> fully unrolled K loop).
//   out[b,co,h,w] = act( sum_{ci,kh,kw} W * in + bias )  (+ res)
// Block: 256 threads (8 wave32), tile = 64 co x 128 px.
// Wave w -> mtile = w&3, ntiles (w>>2)*4..+3.
// Weights: async-copied global->LDS (ASYNCcnt), A frags via ds_load_b128.
// Activations: f32 global -> bf16 LDS (channel-innermost, +4 pad).
// ---------------------------------------------------------------------------
template <int TAPS, int NCC>
__global__ __launch_bounds__(256)
void conv_wmma_kernel(const float* __restrict__ in0, const float* __restrict__ in1,
                      const float* __restrict__ in2,
                      const unsigned short* __restrict__ wfrag,
                      const float* __restrict__ bias,
                      const float* __restrict__ res,
                      float* __restrict__ out,
                      int Cin, int cstride, int Cout, int act) {
    constexpr int HALO  = (TAPS == 9) ? 1 : 0;
    constexpr int R     = (TAPS == 9) ? 3 : 1;
    constexpr int XE    = WTILE + 2 * HALO;
    constexpr int Cc    = NCC * 32;
    constexpr int Cpad  = Cc + 4;                 // 34-dword row stride (bank-safe)
    constexpr int AELEM = ((R * XE * Cpad + 7) & ~7);   // ushorts, 16B aligned
    constexpr int WELEM = TAPS * NCC * 4 * 32 * 16;     // ushorts (exact 2048*TAPS*NCC)

    extern __shared__ unsigned short dynsm[];
    unsigned short* smem = dynsm;                 // activations
    unsigned short* wlds = dynsm + AELEM;         // weight fragments

    const int tid = threadIdx.x;
    const int w0  = blockIdx.x * WTILE;
    const int h   = blockIdx.y;
    const int b   = blockIdx.z;

    // ---- kick off async weight copy: global -> LDS (16B per lane per op) ----
    for (int i = tid * 8; i < WELEM; i += 256 * 8)
        async_g2l_b128(&wlds[i], wfrag + i);

    // ---- stage input tile (f32 -> bf16), channel-innermost, zero halo ----
    constexpr int TOTAL = R * XE * Cc;
    for (int idx = tid; idx < TOTAL; idx += 256) {
        int x  = idx % XE;
        int t2 = idx / XE;
        int r  = t2 % R;
        int ci = t2 / R;
        int gw = w0 - HALO + x;
        int gh = h - HALO + r;
        float v = 0.f;
        if (ci < Cin && (unsigned)gw < (unsigned)IMG_W && (unsigned)gh < (unsigned)IMG_H) {
            int tsel = ci / cstride;
            int cl   = ci - tsel * cstride;
            const float* src = (tsel == 0) ? in0 : ((tsel == 1) ? in1 : in2);
            v = src[((b * cstride + cl) * IMG_H + gh) * IMG_W + gw];
        }
        smem[(r * XE + x) * Cpad + ci] = f2bf(v);
    }

    const int wave = tid >> 5, lane = tid & 31;
    const int mt   = wave & 3;
    const int ntb  = (wave >> 2) * 4;
    const int lh   = lane >> 4;                   // lane half
    const int ln   = lane & 15;

    // prefetch the fused-residual tile while we compute
    if (res) __builtin_prefetch(&res[((b * Cout) * IMG_H + h) * IMG_W + w0 + ntb * 16], 0, 1);

    wait_asynccnt0();                             // weight frags landed in LDS
    __syncthreads();                              // activation staging visible

    v8f zero = {0.f, 0.f, 0.f, 0.f, 0.f, 0.f, 0.f, 0.f};
    v8f acc[4];
#pragma unroll
    for (int q = 0; q < 4; ++q) acc[q] = zero;

#pragma unroll
    for (int t = 0; t < TAPS; ++t) {
        constexpr int dummy = 0; (void)dummy;
        const int kh = (TAPS == 9) ? (t / 3) : 0;
        const int kw = (TAPS == 9) ? (t % 3) : 0;
#pragma unroll
        for (int cc = 0; cc < NCC; ++cc) {
            const int kt = t * NCC + cc;
            FragBF a;
            const uint4* ap = (const uint4*)(wlds + (((kt * 4 + mt) * 32 + lane) << 4));
            a.u[0] = ap[0];
            a.u[1] = ap[1];
            const int cbase = cc * 32 + lh * 8;
#pragma unroll
            for (int q = 0; q < 4; ++q) {
                const int x = (ntb + q) * 16 + ln + kw;   // kw==0 for 1x1
                const unsigned short* sp = smem + (kh * XE + x) * Cpad + cbase;
                FragBF bf;
                const unsigned long long* qp = (const unsigned long long*)sp;
                bf.q[0] = qp[0]; bf.q[1] = qp[1]; bf.q[2] = qp[4]; bf.q[3] = qp[5];
                acc[q] = __builtin_amdgcn_wmma_f32_16x16x32_bf16(
                    false, a.v, false, bf.v, (short)0, acc[q], false, false);
            }
        }
    }

    // ---- epilogue: bias + activation + optional residual, store f32 ----
#pragma unroll
    for (int q = 0; q < 4; ++q) {
        const int wp2 = w0 + (ntb + q) * 16 + ln;
#pragma unroll
        for (int v = 0; v < 8; ++v) {
            const int co = mt * 16 + lh * 8 + v;
            if (co < Cout) {
                float val = acc[q][v] + bias[co];
                if (act == 1)      val = (val >= 0.f) ? val : 0.2f * val;   // LeakyReLU(0.2)
                else if (act == 2) val = fmaxf(val, 0.f);                   // ReLU
                else if (act == 3) val = 1.f / (1.f + expf(-val));          // sigmoid
                int oi = ((b * Cout + co) * IMG_H + h) * IMG_W + wp2;
                if (res) val += res[oi];
                out[oi] = val;
            }
        }
    }
}

// ---------------------------------------------------------------------------
// Channel attention helpers
// ---------------------------------------------------------------------------
__global__ void gap_kernel(const float* __restrict__ x, float* __restrict__ mean) {
    int bc = blockIdx.x;
    const float* p = x + (size_t)bc * HW;
    float s = 0.f;
    for (int i = threadIdx.x; i < HW; i += 256) s += p[i];
    __shared__ float red[256];
    red[threadIdx.x] = s;
    __syncthreads();
    for (int st = 128; st > 0; st >>= 1) {
        if (threadIdx.x < st) red[threadIdx.x] += red[threadIdx.x + st];
        __syncthreads();
    }
    if (threadIdx.x == 0) mean[bc] = red[0] * (1.0f / HW);
}

__global__ void ca_mlp_kernel(const float* __restrict__ mean,
                              const float* __restrict__ wd, const float* __restrict__ bd,
                              const float* __restrict__ wu, const float* __restrict__ bu,
                              float* __restrict__ scale) {
    int b = blockIdx.x;
    int c = threadIdx.x;                       // 0..63
    __shared__ float y1[4];
    if (c < 4) {
        float s = bd[c];
        for (int k = 0; k < 64; ++k) s += wd[c * 64 + k] * mean[b * 64 + k];
        y1[c] = fmaxf(s, 0.f);
    }
    __syncthreads();
    float s = bu[c];
    for (int j = 0; j < 4; ++j) s += wu[c * 4 + j] * y1[j];
    scale[b * 64 + c] = 1.f / (1.f + expf(-s));
}

__global__ void scale_add_kernel(const float* __restrict__ x, const float* __restrict__ sc,
                                 const float* __restrict__ skip, float* __restrict__ out) {
    int i = blockIdx.x * 256 + threadIdx.x;
    int bc = i >> 16;                          // HW == 65536
    out[i] = x[i] * sc[bc] + skip[i];
}

// gk*(S+1)
__global__ void kmul1p_kernel(const float* __restrict__ k, const float* __restrict__ S,
                              float* __restrict__ out) {
    int i = blockIdx.x * 256 + threadIdx.x;
    out[i] = k[i] * (S[i] + 1.0f);
}

// fk = tanh(k)/(aff+1e-8); s = max(sum|fk| + 1e-4, 1); out = fk/s   (per pixel, 9 ch)
__global__ void getkernel_kernel(const float* __restrict__ k, const float* __restrict__ aff,
                                 float* __restrict__ out) {
    int i = blockIdx.x * 256 + threadIdx.x;    // over NB*HW
    int b = i / HW;
    int p = i - b * HW;
    float inv = 1.f / (aff[0] + 1e-8f);
    float f[9];
    float s = 0.f;
#pragma unroll
    for (int j = 0; j < 9; ++j) {
        f[j] = tanhf(k[(b * 9 + j) * HW + p]) * inv;
        s += fabsf(f[j]);
    }
    s += 1e-4f;
    s = fmaxf(s, 1.f);
    float rs = 1.f / s;
#pragma unroll
    for (int j = 0; j < 9; ++j) out[(b * 9 + j) * HW + p] = f[j] * rs;
}

// out[b,c,h,w] = sum_t gk[b,t,h,w] * xpad[b,c,h+t/3-1,w+t%3-1] + x[b,c,h,w]
__global__ void kernagg_kernel(const float* __restrict__ gk, const float* __restrict__ x,
                               float* __restrict__ out) {
    const int w = threadIdx.x;
    const int h = blockIdx.x;
    const int b = blockIdx.y;
    float g[9];
#pragma unroll
    for (int t = 0; t < 9; ++t) g[t] = gk[((b * 9 + t) * IMG_H + h) * IMG_W + w];
    for (int c = 0; c < NCH; ++c) {
        const float* xp = x + (size_t)(b * NCH + c) * HW;
        float s = xp[h * IMG_W + w];
#pragma unroll
        for (int t = 0; t < 9; ++t) {
            int hh = h + t / 3 - 1;
            int ww = w + t % 3 - 1;
            if ((unsigned)hh < (unsigned)IMG_H && (unsigned)ww < (unsigned)IMG_W)
                s += g[t] * xp[hh * IMG_W + ww];
        }
        out[((b * NCH + c) * IMG_H + h) * IMG_W + w] = s;
    }
}

// ---------------------------------------------------------------------------
// Host orchestration
// ---------------------------------------------------------------------------
extern "C" void kernel_launch(void* const* d_in, const int* in_sizes, int n_in,
                              void* d_out, int out_size, void* d_ws, size_t ws_size,
                              hipStream_t stream) {
    (void)in_sizes; (void)n_in; (void)out_size; (void)ws_size;

    const float* depth = (const float*)d_in[0];
    const float* image = (const float*)d_in[1];
    const float* seg   = (const float*)d_in[2];
    const float* ns    = (const float*)d_in[3];
    const float* S_seg = (const float*)d_in[4];
    const float* S_ns  = (const float*)d_in[5];
    const float* S_rgb = (const float*)d_in[6];
    const float* P     = (const float*)d_in[7];
    float* outp = (float*)d_out;

    // ---- param offsets: jax pytree leaf order (keys sorted; 'b' before 'w') ----
    struct CP { long b, w; int cin, cout, taps; };
    long off = 0;
    auto cp = [&](int cin, int cout, int k) -> CP {
        CP r; r.cin = cin; r.cout = cout; r.taps = k * k;
        r.b = off; off += cout;
        r.w = off; off += (long)cout * cin * k * k;
        return r;
    };
    struct RGP { CP c1, c2, down, up, tail; };
    auto rgp = [&]() -> RGP {
        RGP r;
        r.c1 = cp(64, 64, 3); r.c2 = cp(64, 64, 3);
        r.down = cp(64, 4, 1); r.up = cp(4, 64, 1);
        r.tail = cp(64, 64, 3);
        return r;
    };
    struct MGFP { long aff; CP d1, d2, d3, dk1, dk2, g1, g2, g3, gk1, gk2; RGP gc, ic; };
    auto mgfp = [&]() -> MGFP {
        MGFP m;
        m.aff = off; off += 1;
        m.d1 = cp(64, 64, 3); m.d2 = cp(9, 9, 1); m.d3 = cp(64, 64, 3);
        m.dk1 = cp(64, 64, 1); m.dk2 = cp(64, 9, 1);
        m.g1 = cp(64, 64, 3); m.g2 = cp(9, 9, 1); m.g3 = cp(64, 64, 3);
        m.gc = rgp();
        m.gk1 = cp(64, 64, 1); m.gk2 = cp(64, 9, 1);
        m.ic = rgp();
        return m;
    };
    // top-level sorted: c1,c2,c3,c4,conv11,conv12,conv21,conv22,conv31,w1,w2,w3
    CP c1 = cp(64, 64, 3), c2 = cp(64, 64, 3), c3 = cp(64, 64, 3), c4 = cp(64, 64, 3);
    CP conv11 = cp(64, 64, 3), conv12 = cp(64, 64, 3), conv21 = cp(64, 64, 3);
    CP conv22 = cp(192, 64, 1), conv31 = cp(64, 1, 3);
    MGFP w1 = mgfp(), w2 = mgfp(), w3 = mgfp();

    // ---- workspace bump allocator ----
    char* wsbase = (char*)d_ws;
    size_t wo = 0;
    auto alloc = [&](size_t bytes) -> void* {
        void* p = wsbase + wo;
        wo += (bytes + 255) & ~(size_t)255;
        return p;
    };
    const size_t T  = (size_t)NB * NCH * HW;   // 8,388,608 floats
    const size_t T9 = (size_t)NB * 9 * HW;
    auto allocT = [&]() { return (float*)alloc(T * 4); };
    auto allocK = [&]() { return (float*)alloc(T9 * 4); };

    // ---- weight packing ----
    struct ConvW { CP p; const unsigned short* frag; };
    auto mk = [&](const CP& c) -> ConvW {
        int nCc = (c.cin + 31) / 32;
        size_t elems = (size_t)c.taps * nCc * 4 * 32 * 16;
        unsigned short* frag = (unsigned short*)alloc(elems * 2);
        int grid = (int)((elems + 255) / 256);
        pack_kernel<<<grid, 256, 0, stream>>>(P + c.w, frag, c.cin, c.cout, c.taps);
        return ConvW{c, frag};
    };
    struct RGW { ConvW c1, c2, tail; CP down, up; };
    auto mkrg = [&](const RGP& r) -> RGW {
        return RGW{mk(r.c1), mk(r.c2), mk(r.tail), r.down, r.up};
    };
    struct MGFW { const MGFP* p; ConvW d1, g1, d3, g3, dk1, dk2, gk1, gk2, d2, g2; RGW ic, gc; };
    auto mkmgf = [&](const MGFP& m) -> MGFW {
        MGFW w; w.p = &m;
        w.d1 = mk(m.d1); w.g1 = mk(m.g1); w.d3 = mk(m.d3); w.g3 = mk(m.g3);
        w.dk1 = mk(m.dk1); w.dk2 = mk(m.dk2); w.gk1 = mk(m.gk1); w.gk2 = mk(m.gk2);
        w.d2 = mk(m.d2); w.g2 = mk(m.g2);
        w.ic = mkrg(m.ic); w.gc = mkrg(m.gc);
        return w;
    };

    // ---- conv launcher: dispatch to the right template instantiation ----
    auto convN = [&](const ConvW& cw, const float* i0, const float* i1, const float* i2,
                     int cstride, const float* resid, float* dst, int act) {
        int taps = cw.p.taps;
        int nCc  = (cw.p.cin + 31) / 32;
        int R  = (taps == 9) ? 3 : 1;
        int XE = WTILE + ((taps == 9) ? 2 : 0);
        size_t aelem = ((size_t)R * XE * (nCc * 32 + 4) + 7) & ~(size_t)7;
        size_t welem = (size_t)taps * nCc * 2048;
        size_t sm = (aelem + welem) * 2;
        dim3 grid(IMG_W / WTILE, IMG_H, NB);
        if (taps == 9)
            conv_wmma_kernel<9, 2><<<grid, 256, sm, stream>>>(
                i0, i1, i2, cw.frag, P + cw.p.b, resid, dst, cw.p.cin, cstride, cw.p.cout, act);
        else if (nCc == 6)
            conv_wmma_kernel<1, 6><<<grid, 256, sm, stream>>>(
                i0, i1, i2, cw.frag, P + cw.p.b, resid, dst, cw.p.cin, cstride, cw.p.cout, act);
        else if (nCc == 2)
            conv_wmma_kernel<1, 2><<<grid, 256, sm, stream>>>(
                i0, i1, i2, cw.frag, P + cw.p.b, resid, dst, cw.p.cin, cstride, cw.p.cout, act);
        else
            conv_wmma_kernel<1, 1><<<grid, 256, sm, stream>>>(
                i0, i1, i2, cw.frag, P + cw.p.b, resid, dst, cw.p.cin, cstride, cw.p.cout, act);
    };
    auto conv1in = [&](const ConvW& cw, const float* i0, const float* resid, float* dst, int act) {
        convN(cw, i0, nullptr, nullptr, cw.p.cin, resid, dst, act);
    };

    float* means  = (float*)alloc(NB * NCH * 4);
    float* scales = (float*)alloc(NB * NCH * 4);

    // ResidualGroup: dst = conv(tail, rcab(x)) + x
    auto runRG = [&](const RGW& rw, const float* x, float* dst, float* t1, float* t2) {
        conv1in(rw.c1, x, nullptr, t1, 1);                        // LeakyReLU(0.2)
        conv1in(rw.c2, t1, nullptr, t2, 0);
        gap_kernel<<<NB * NCH, 256, 0, stream>>>(t2, means);
        ca_mlp_kernel<<<NB, 64, 0, stream>>>(means, P + rw.down.w, P + rw.down.b,
                                             P + rw.up.w, P + rw.up.b, scales);
        scale_add_kernel<<<(int)(T / 256), 256, 0, stream>>>(t2, scales, x, t1);
        conv1in(rw.tail, t1, x, dst, 0);                          // + x fused
    };

    // temps + persistents
    float* dC = allocT(); float* iC = allocT(); float* sC = allocT(); float* nC = allocT();
    float* drsn = allocT(); float* drs = allocT();
    float* bA = allocT(); float* bB = allocT(); float* bC = allocT();
    float* bD = allocT(); float* bE = allocT(); float* bF = allocT();
    float* mo = allocT();
    float* k1 = allocK(); float* k2 = allocK();

    auto runMGF = [&](const MGFW& mw, const float* din, const float* gin, const float* S,
                      const float* resA, float* outA, float* outB) {
        conv1in(mw.d1, din, nullptr, bA, 0);                      // dd
        conv1in(mw.g1, gin, nullptr, bB, 0);                      // gg
        runRG(mw.ic, bA, bC, bD, bE);                             // inputs_depth -> bC
        conv1in(mw.gk1, bB, nullptr, bD, 2);                      // relu
        conv1in(mw.gk2, bD, nullptr, k1, 0);                      // gk [B,9,H,W]
        kmul1p_kernel<<<(int)(T9 / 256), 256, 0, stream>>>(k1, S, k2);
        conv1in(mw.g2, k2, nullptr, k1, 0);                       // 9 -> 9
        getkernel_kernel<<<(NB * HW) / 256, 256, 0, stream>>>(k1, P + mw.p->aff, k2);
        kernagg_kernel<<<dim3(IMG_H, NB), 256, 0, stream>>>(k2, bC, bF);   // w_depth -> bF
        runRG(mw.gc, bB, bC, bD, bE);                             // inputs_guide -> bC
        conv1in(mw.dk1, bF, nullptr, bD, 2);
        conv1in(mw.dk2, bD, nullptr, k1, 0);
        kmul1p_kernel<<<(int)(T9 / 256), 256, 0, stream>>>(k1, S, k2);
        conv1in(mw.d2, k2, nullptr, k1, 0);
        getkernel_kernel<<<(NB * HW) / 256, 256, 0, stream>>>(k1, P + mw.p->aff, k2);
        kernagg_kernel<<<dim3(IMG_H, NB), 256, 0, stream>>>(k2, bC, bB);   // w_guide -> bB
        conv1in(mw.d3, bF, resA, outA, 0);                        // + resA fused
        conv1in(mw.g3, bB, nullptr, outB, 0);
    };

    // ---- pack all weights first (stream-ordered before use) ----
    ConvW wc1 = mk(c1), wc2 = mk(c2), wc3 = mk(c3), wc4 = mk(c4);
    ConvW w11 = mk(conv11), w12 = mk(conv12), w21 = mk(conv21);
    ConvW w22 = mk(conv22), w31 = mk(conv31);
    MGFW m1 = mkmgf(w1), m2 = mkmgf(w2), m3 = mkmgf(w3);

    // ---- full pipeline ----
    float* out_rgb = outp + (size_t)NB * HW;                      // NB*1*H*W head
    float* out_seg = out_rgb + T;
    float* out_ns  = out_seg + T;

    conv1in(wc1, depth, nullptr, dC, 0);
    conv1in(wc2, image, nullptr, iC, 0);
    conv1in(wc3, seg,   nullptr, sC, 0);
    conv1in(wc4, ns,    nullptr, nC, 0);

    runMGF(m2, dC, nC, S_ns, dC, mo, out_ns);                     // mo = d3(w_depth)+depth
    conv1in(w12, mo, nullptr, drsn, 0);                           // out_drsn
    runMGF(m3, drsn, sC, S_seg, drsn, mo, out_seg);
    conv1in(w21, mo, nullptr, drs, 0);                            // out_drs
    runMGF(m1, drs, iC, S_rgb, drs, mo, out_rgb);
    float* odr = nC;                                              // reuse (ns_c dead)
    conv1in(w11, mo, nullptr, odr, 0);                            // out_dr
    convN(w22, odr, drsn, drs, 64, dC, bA, 0);                    // conv22(cat)+depth -> out0
    conv1in(w31, bA, nullptr, outp, 0);                           // final C->1 conv
}
